// LSTM_13915694039291
// MI455X (gfx1250) — matrix operand order repaired and checked by reading, not compiled, not run on hardware
//
#include <hip/hip_runtime.h>
#include <hip/hip_bf16.h>

// ---------------------------------------------------------------------------
// Fully fused 2-layer LSTM for MI455X (gfx1250, wave32, WMMA).
// One workgroup = 16 batch rows, runs all 1024 timesteps of both layers.
// Recurrent + input GEMMs via v_wmma_f32_16x16x32_bf16 (fp32 accumulate).
// ---------------------------------------------------------------------------

typedef __attribute__((ext_vector_type(16))) __bf16 v16bf;
typedef __attribute__((ext_vector_type(8)))  __bf16 v8bf;
typedef __attribute__((ext_vector_type(8)))  float  v8f;

#define HDIM   128
#define GDIM   512          // 4*H
#define BSZ    256
#define TLEN   1024
#define MROWS  16           // batch rows per workgroup
#define HSTRIDE 136         // padded bf16 row stride for h state in LDS

// LDS layout (bytes):
//   whh0 fragments : 128 frags * 32 lanes * 16 bf16 = 131072
//   whh1 fragments : 131072
//   h0 state (x2 buffers): 2*16*HSTRIDE*2 = 8704
//   h1 state (x2 buffers): 8704
//   h1 final fp32  : 16*128*4 = 8192
#define OFF_WHH0  0
#define OFF_WHH1  131072
#define OFF_H0    262144
#define OFF_H1    (OFF_H0 + 2*MROWS*HSTRIDE*2)
#define OFF_H1F   (OFF_H1 + 2*MROWS*HSTRIDE*2)
#define SMEM_BYTES (OFF_H1F + MROWS*HDIM*4)

__device__ __forceinline__ float sigmoidf_(float v) {
    return 1.0f / (1.0f + __expf(-v));
}

// A fragment (16x32 bf16, M = batch row) from plain [m][k] LDS rows.
// Contiguous layout: lanes 0-15: e0-15=K+0..15 ; lanes 16-31: e0-15=K+16..31.
__device__ __forceinline__ v16bf load_a_frag(const __bf16* hbase, int kt, int lr, int lh) {
    const __bf16* row = hbase + lr * HSTRIDE;
    int k0 = kt * 32 + lh * 16;
    v8bf lo = *(const v8bf*)(row + k0);
    v8bf hi = *(const v8bf*)(row + k0 + 8);
    v16bf r;
#pragma unroll
    for (int e = 0; e < 8; ++e) { r[e] = lo[e]; r[e + 8] = hi[e]; }
    return r;
}

// B fragment pre-swizzled in LDS: frag block = 32 lanes * 16 bf16 contiguous.
__device__ __forceinline__ v16bf load_b_frag(const __bf16* fragbase, int lane) {
    const __bf16* p = fragbase + lane * 16;
    v8bf lo = *(const v8bf*)p;
    v8bf hi = *(const v8bf*)(p + 8);
    v16bf r;
#pragma unroll
    for (int e = 0; e < 8; ++e) { r[e] = lo[e]; r[e + 8] = hi[e]; }
    return r;
}

__global__ void __launch_bounds__(256, 1)
lstm2_fused(const float* __restrict__ x,      // (B,T,2)
            const float* __restrict__ h0in,   // (2,B,H)
            const float* __restrict__ c0in,   // (2,B,H)
            const float* __restrict__ Wih0,   // (512,2)
            const float* __restrict__ Whh0,   // (512,128)
            const float* __restrict__ bih0,
            const float* __restrict__ bhh0,
            const float* __restrict__ Wih1,   // (512,128)
            const float* __restrict__ Whh1,   // (512,128)
            const float* __restrict__ bih1,
            const float* __restrict__ bhh1,
            const float* __restrict__ Wbw,    // (1,128)
            const float* __restrict__ bbw,
            const float* __restrict__ Wcls,   // (1,128)
            const float* __restrict__ bcls,
            float* __restrict__ out)
{
    extern __shared__ char smem[];
    __bf16* whh0f = (__bf16*)(smem + OFF_WHH0);
    __bf16* whh1f = (__bf16*)(smem + OFF_WHH1);
    __bf16* h0buf = (__bf16*)(smem + OFF_H0);   // [2][16][HSTRIDE]
    __bf16* h1buf = (__bf16*)(smem + OFF_H1);
    float*  h1f   = (float*)(smem + OFF_H1F);   // [16][128] fp32

    const int tid  = threadIdx.x;
    const int lane = tid & 31;
    const int wv   = tid >> 5;        // wave 0..7 -> hidden slice
    const int m0   = blockIdx.x * MROWS;
    const int lr   = lane & 15;
    const int lh   = lane >> 4;       // 0/1
    const int jloc = wv * 16 + lr;    // hidden index this lane owns

    // ---- one-time: swizzle W_hh0 / W_hh1 into B-fragment layout in LDS ----
    // frag id fid = nt*4+kt ; element e -> W[g = nt*16 + (ln&15)][k = kt*32 + 16*(ln>>4) + e]
    for (int idx = tid; idx < 4096; idx += 256) {
        int ln  = idx & 31;
        int fid = idx >> 5;
        int kt  = fid & 3;
        int nt  = fid >> 2;
        int g   = nt * 16 + (ln & 15);
        int kb  = kt * 32 + ((ln >> 4) << 4);
        const float* s0 = Whh0 + g * HDIM + kb;
        const float* s1 = Whh1 + g * HDIM + kb;
        __bf16* d0 = whh0f + (fid * 32 + ln) * 16;
        __bf16* d1 = whh1f + (fid * 32 + ln) * 16;
#pragma unroll
        for (int e = 0; e < 16; ++e) { d0[e] = (__bf16)s0[e]; d1[e] = (__bf16)s1[e]; }
    }
    // init h state (buffer 0)
    for (int idx = tid; idx < MROWS * HDIM; idx += 256) {
        int m = idx >> 7;
        int k = idx & 127;
        h0buf[m * HSTRIDE + k] = (__bf16)h0in[(0 * BSZ + m0 + m) * HDIM + k];
        h1buf[m * HSTRIDE + k] = (__bf16)h0in[(1 * BSZ + m0 + m) * HDIM + k];
    }

    // ---- per-lane W_ih1 fragments kept in VGPRs (16 frags * 8 VGPRs) ----
    v16bf wih1f[4][4];
#pragma unroll
    for (int gg = 0; gg < 4; ++gg) {
#pragma unroll
        for (int kt = 0; kt < 4; ++kt) {
            int nt = gg * 8 + wv;
            int g  = nt * 16 + lr;
            int kb = kt * 32 + (lh << 4);
            const float* s = Wih1 + g * HDIM + kb;
            v16bf v;
#pragma unroll
            for (int e = 0; e < 16; ++e) v[e] = (__bf16)s[e];
            wih1f[gg][kt] = v;
        }
    }

    // per-lane gate constants (one per gate for hidden unit jloc)
    float wi0g[4], wi1g[4], b0g[4], b1g[4];
#pragma unroll
    for (int gg = 0; gg < 4; ++gg) {
        int n = gg * HDIM + jloc;
        wi0g[gg] = Wih0[n * 2 + 0];
        wi1g[gg] = Wih0[n * 2 + 1];
        b0g[gg]  = bih0[n] + bhh0[n];
        b1g[gg]  = bih1[n] + bhh1[n];
    }

    // cell state registers: element r <-> (m = r + 8*lh, hidden = jloc)
    float c0r[8], c1r[8], h0v[8], h1v[8];
#pragma unroll
    for (int r = 0; r < 8; ++r) {
        int b = m0 + r + 8 * lh;
        c0r[r] = c0in[(0 * BSZ + b) * HDIM + jloc];
        c1r[r] = c0in[(1 * BSZ + b) * HDIM + jloc];
        h0v[r] = 0.f; h1v[r] = 0.f;
    }

    __syncthreads();

    for (int t = 0; t < TLEN; ++t) {
        const int p = t & 1;          // read buffer
        const int q = p ^ 1;          // write buffer
        __bf16* h0r = h0buf + p * MROWS * HSTRIDE;
        __bf16* h0w = h0buf + q * MROWS * HSTRIDE;
        __bf16* h1r = h1buf + p * MROWS * HSTRIDE;
        __bf16* h1w = h1buf + q * MROWS * HSTRIDE;

        // ---------------- phase 1: read old state fragments ----------------
        v16bf a0[4], ah1[4];
#pragma unroll
        for (int kt = 0; kt < 4; ++kt) {
            a0[kt]  = load_a_frag(h0r, kt, lr, lh);
            ah1[kt] = load_a_frag(h1r, kt, lr, lh);
        }

        // ---------------- layer 0 ----------------
        float x0[8], x1[8];
#pragma unroll
        for (int r = 0; r < 8; ++r) {
            int b = m0 + r + 8 * lh;
            const float* xp = x + ((size_t)b * TLEN + t) * 2;
            x0[r] = xp[0]; x1[r] = xp[1];
            if (t + 1 < TLEN) __builtin_prefetch(xp + 2, 0, 1);
        }
        v8f acc[4];
#pragma unroll
        for (int gg = 0; gg < 4; ++gg) {
            v8f a;
#pragma unroll
            for (int r = 0; r < 8; ++r)
                a[r] = fmaf(x0[r], wi0g[gg], fmaf(x1[r], wi1g[gg], b0g[gg]));
            acc[gg] = a;
        }
#pragma unroll
        for (int gg = 0; gg < 4; ++gg) {
            int nt = gg * 8 + wv;
#pragma unroll
            for (int kt = 0; kt < 4; ++kt) {
                v16bf bf = load_b_frag(whh0f + ((nt * 4 + kt) * 32) * 16, lane);
                acc[gg] = __builtin_amdgcn_wmma_f32_16x16x32_bf16(
                    false, a0[kt], false, bf, (short)0, acc[gg], false, false);
            }
        }
#pragma unroll
        for (int r = 0; r < 8; ++r) {
            float iv = sigmoidf_(acc[0][r]);
            float fv = sigmoidf_(acc[1][r]);
            float gv = tanhf(acc[2][r]);
            float ov = sigmoidf_(acc[3][r]);
            float c  = fv * c0r[r] + iv * gv;
            c0r[r]   = c;
            h0v[r]   = ov * tanhf(c);
            h0w[(r + 8 * lh) * HSTRIDE + jloc] = (__bf16)h0v[r];
        }

        __syncthreads();   // h0(new) visible; all old-state reads complete

        // ---------------- layer 1 ----------------
        v16bf a1[4];
#pragma unroll
        for (int kt = 0; kt < 4; ++kt) a1[kt] = load_a_frag(h0w, kt, lr, lh);

        v8f acc1[4];
#pragma unroll
        for (int gg = 0; gg < 4; ++gg) {
            v8f a;
#pragma unroll
            for (int r = 0; r < 8; ++r) a[r] = b1g[gg];
            acc1[gg] = a;
        }
#pragma unroll
        for (int gg = 0; gg < 4; ++gg) {
            // input contribution: y0_t @ W_ih1^T  (B frags live in registers)
#pragma unroll
            for (int kt = 0; kt < 4; ++kt)
                acc1[gg] = __builtin_amdgcn_wmma_f32_16x16x32_bf16(
                    false, a1[kt], false, wih1f[gg][kt], (short)0, acc1[gg], false, false);
            // recurrent contribution: h1 @ W_hh1^T
            int nt = gg * 8 + wv;
#pragma unroll
            for (int kt = 0; kt < 4; ++kt) {
                v16bf bf = load_b_frag(whh1f + ((nt * 4 + kt) * 32) * 16, lane);
                acc1[gg] = __builtin_amdgcn_wmma_f32_16x16x32_bf16(
                    false, ah1[kt], false, bf, (short)0, acc1[gg], false, false);
            }
        }
#pragma unroll
        for (int r = 0; r < 8; ++r) {
            float iv = sigmoidf_(acc1[0][r]);
            float fv = sigmoidf_(acc1[1][r]);
            float gv = tanhf(acc1[2][r]);
            float ov = sigmoidf_(acc1[3][r]);
            float c  = fv * c1r[r] + iv * gv;
            c1r[r]   = c;
            h1v[r]   = ov * tanhf(c);
            h1w[(r + 8 * lh) * HSTRIDE + jloc] = (__bf16)h1v[r];
        }

        __syncthreads();   // h1(new) visible before next iteration reads it
    }

    // ---- epilogue: h_n / c_n from registers; heads via fp32 LDS buffer ----
    // out layout: [bw 256 | cls 256 | h_n 2*256*128 | c_n 2*256*128]
#pragma unroll
    for (int r = 0; r < 8; ++r) {
        int b = m0 + r + 8 * lh;
        out[512 +          0 + b * HDIM + jloc] = h0v[r];
        out[512 +      32768 + b * HDIM + jloc] = h1v[r];
        out[66048 +        0 + b * HDIM + jloc] = c0r[r];
        out[66048 +    32768 + b * HDIM + jloc] = c1r[r];
        h1f[(r + 8 * lh) * HDIM + jloc] = h1v[r];
    }
    __syncthreads();
    if (tid < MROWS) {
        float s1 = 0.f, s2 = 0.f;
        for (int k = 0; k < HDIM; ++k) {
            float hv = h1f[tid * HDIM + k];
            s1 = fmaf(hv, Wbw[k], s1);
            s2 = fmaf(hv, Wcls[k], s2);
        }
        out[m0 + tid]       = s1 + bbw[0];
        out[256 + m0 + tid] = s2 + bcls[0];
    }
}

extern "C" void kernel_launch(void* const* d_in, const int* in_sizes, int n_in,
                              void* d_out, int out_size, void* d_ws, size_t ws_size,
                              hipStream_t stream) {
    (void)in_sizes; (void)n_in; (void)out_size; (void)d_ws; (void)ws_size;
    const float* x    = (const float*)d_in[0];
    const float* h0   = (const float*)d_in[1];
    const float* c0   = (const float*)d_in[2];
    const float* Wih0 = (const float*)d_in[3];
    const float* Whh0 = (const float*)d_in[4];
    const float* bih0 = (const float*)d_in[5];
    const float* bhh0 = (const float*)d_in[6];
    const float* Wih1 = (const float*)d_in[7];
    const float* Whh1 = (const float*)d_in[8];
    const float* bih1 = (const float*)d_in[9];
    const float* bhh1 = (const float*)d_in[10];
    const float* Wbw  = (const float*)d_in[11];
    const float* bbw  = (const float*)d_in[12];
    const float* Wcls = (const float*)d_in[13];
    const float* bcls = (const float*)d_in[14];
    float* out = (float*)d_out;

    // 281 KB dynamic LDS (< 320 KB/WGP on CDNA5); raise the runtime cap.
    hipFuncSetAttribute((const void*)lstm2_fused,
                        hipFuncAttributeMaxDynamicSharedMemorySize,
                        (int)SMEM_BYTES);

    hipLaunchKernelGGL(lstm2_fused, dim3(BSZ / MROWS), dim3(256), SMEM_BYTES, stream,
                       x, h0, c0, Wih0, Whh0, bih0, bhh0,
                       Wih1, Whh1, bih1, bhh1, Wbw, bbw, Wcls, bcls, out);
}
